// MultiHeadedAttention_57732950393363
// MI455X (gfx1250) — matrix operand order repaired and checked by reading, not compiled, not run on hardware
//
#include <hip/hip_runtime.h>
#include <hip/hip_bf16.h>

// ---------------------------------------------------------------------------
// MultiHeadedAttention for MI455X (gfx1250, wave32, WMMA bf16 16x16x32)
//   B=4, S=2048, H=16, D_K=64, D_MODEL=1024
// Pipeline:
//   1) proj_split_heads x3 : Q/K = X @ W^T + b  -> bf16 [B,H,S,64]
//                            V                  -> bf16 [B,H,64,S] (transposed
//                            so P*V B-tiles are contiguous loads)
//   2) flash_attn          : online-softmax attention; K/V/src_mask tiles all
//                            staged into LDS by the Tensor Data Mover
//                            (tensor_load_to_lds, double buffered,
//                            s_wait_tensorcnt) -> inner loop is ds+wmma only
//   3) proj_out            : out = A @ Wo^T + bo   (bf16 in -> f32 out)
// ---------------------------------------------------------------------------

#define B_ 4
#define S_ 2048
#define H_ 16
#define DK_ 64
#define DM_ 1024
#define M_ (B_ * S_)          // 8192 rows in the flattened activations

typedef __bf16 bf16_t;
typedef __attribute__((ext_vector_type(16))) __bf16 v16bf;
typedef __attribute__((ext_vector_type(8)))  float  v8f;
typedef __attribute__((ext_vector_type(4)))  unsigned int v4u;
typedef __attribute__((ext_vector_type(8)))  unsigned int v8u;

__device__ __forceinline__ v8f wmma_bf16(v16bf a, v16bf b, v8f c) {
  // D = A(16x32) * B(32x16) + C(16x16), f32 accumulate
  return __builtin_amdgcn_wmma_f32_16x16x32_bf16(
      /*neg_a=*/false, a, /*neg_b=*/false, b,
      /*c_mod=*/(short)0, c, /*reuse_a=*/false, /*reuse_b=*/false);
}

// ---------------------------------------------------------------------------
// Tensor Data Mover: 2-D tile load Global -> LDS (CDNA5 ISA)
// D# group0: count=1 | lds_addr | global_addr[56:0] | type=2
// D# group1: data_size | tensor_dim0/1 | tile_dim0/1 | dim0 stride
// Issued via inline asm (portable across toolchains); tracked by TENSORcnt.
// All arguments are wave-uniform. dsize_code: 0=1B,1=2B,2=4B,3=8B.
// ---------------------------------------------------------------------------
__device__ __forceinline__ void tdm_load_2d(unsigned int lds_addr,
                                            const void* gptr,
                                            unsigned int dsize_code,
                                            unsigned int tensor_d0,  // row len (elems)
                                            unsigned int tensor_d1,  // #rows
                                            unsigned int tile_d0,    // tile row len
                                            unsigned int tile_d1,    // tile #rows
                                            unsigned int stride0)    // row stride (elems)
{
  const unsigned long long ga = (unsigned long long)gptr;
  v4u g0;
  g0[0] = 1u;                                   // count=1 (valid descriptor)
  g0[1] = lds_addr;                             // LDS byte address
  g0[2] = (unsigned int)ga;                     // global_addr[31:0]
  g0[3] = ((unsigned int)(ga >> 32) & 0x01FFFFFFu) // global_addr[56:32]
          | 0x80000000u;                        // type=2 ("image") at [127:126]
  v8u g1 = {0u, 0u, 0u, 0u, 0u, 0u, 0u, 0u};
  g1[0] = (dsize_code << 16);                   // data_size
  g1[1] = (tensor_d0 & 0xFFFFu) << 16;          // tensor_dim0[15:0]
  g1[2] = (tensor_d0 >> 16) | ((tensor_d1 & 0xFFFFu) << 16);
  g1[3] = (tensor_d1 >> 16) | (tile_d0 << 16);  // tile_dim0
  g1[4] = (tile_d1 & 0xFFFFu);                  // tile_dim1 (tile_dim2=0)
  g1[5] = stride0;                              // tensor_dim0_stride[31:0]
  asm volatile("tensor_load_to_lds %0, %1" :: "s"(g0), "s"(g1) : "memory");
}

__device__ __forceinline__ unsigned int lds_addr_of(const void* p) {
  // generic LDS pointer: low 32 bits are the LDS offset (ISA flat aperture rule)
  return (unsigned int)(unsigned long long)p;
}

// ---- A-matrix (16x32, row-major source, lane=M, vgpr-halves split K) -------
__device__ __forceinline__ v16bf load_a_bf16(const bf16_t* base, int ld) {
  const int lane = threadIdx.x & 31;
  const int m    = lane & 15;
  const int kh   = (lane >> 4) << 4;         // 0 or 16
  const bf16_t* p = base + (size_t)m * ld + kh;
  v16bf a;
#pragma unroll
  for (int i = 0; i < 8; ++i) { a[i] = p[i]; a[i + 8] = p[i + 8]; }
  return a;
}

__device__ __forceinline__ v16bf load_a_f32(const float* __restrict__ base, int ld) {
  const int lane = threadIdx.x & 31;
  const int m    = lane & 15;
  const int kh   = (lane >> 4) << 4;         // 0 or 16
  const float* p = base + (size_t)m * ld + kh;
  v16bf a;
#pragma unroll
  for (int i = 0; i < 8; ++i) {
    a[i]     = (bf16_t)p[i];
    a[i + 8] = (bf16_t)p[i + 8];
  }
  return a;
}

// ---- B-matrix for C = A @ Wt^T : B[k][n] = Wt[n][k], lane=n, contiguous k --
__device__ __forceinline__ v16bf load_bT_f32(const float* __restrict__ base, int ld) {
  const int lane = threadIdx.x & 31;
  const int n    = lane & 15;
  const int kh   = (lane >> 4) << 4;         // 0 or 16
  const float* p = base + (size_t)n * ld + kh;
  v16bf b;
#pragma unroll
  for (int i = 0; i < 16; ++i) b[i] = (bf16_t)p[i];
  return b;
}

__device__ __forceinline__ v16bf load_bT_bf16(const bf16_t* base, int ld) {
  const int lane = threadIdx.x & 31;
  const int n    = lane & 15;
  const int kh   = (lane >> 4) << 4;         // 0 or 16
  const bf16_t* p = base + (size_t)n * ld + kh;
  v16bf b;
#pragma unroll
  for (int i = 0; i < 16; ++i) b[i] = p[i];
  return b;
}

// ---------------------------------------------------------------------------
// Kernel 1: C = X @ W^T + bias -> bf16, head-split.
//   TRANS_OUT=false : Out[b][h][s][d]   (Q, K)
//   TRANS_OUT=true  : Out[b][h][d][s]   (V, so P*V B-tiles are contiguous)
// grid = (M/16)*(DM/64) = 8192 waves, one wave per 16x64 tile
// ---------------------------------------------------------------------------
template <bool TRANS_OUT>
__global__ __launch_bounds__(32) void proj_split_heads(
    const float* __restrict__ X,     // [M_, DM_]
    const float* __restrict__ W,     // [DM_, DM_] (row = out feature)
    const float* __restrict__ bias,  // [DM_]
    bf16_t* __restrict__ Out)
{
  const int tile = blockIdx.x;
  const int m0 = (tile >> 4) * 16;        // DM_/64 = 16 tiles along N
  const int n0 = (tile & 15) * 64;

  v8f acc[4] = {};
  for (int k0 = 0; k0 < DM_; k0 += 32) {
    v16bf a = load_a_f32(X + (size_t)m0 * DM_ + k0, DM_);
#pragma unroll
    for (int j = 0; j < 4; ++j) {
      v16bf bt = load_bT_f32(W + (size_t)(n0 + j * 16) * DM_ + k0, DM_);
      acc[j] = wmma_bf16(a, bt, acc[j]);
    }
  }

  const int lane  = threadIdx.x & 31;
  const int nlo   = lane & 15;
  const int rbase = (lane >> 4) << 3;      // 0 or 8
#pragma unroll
  for (int j = 0; j < 4; ++j) {
    const int n  = n0 + j * 16 + nlo;
    const int h  = n >> 6;                 // /64
    const int d  = n & 63;
    const float bb = bias[n];
#pragma unroll
    for (int r = 0; r < 8; ++r) {
      const int m = m0 + rbase + r;
      const int b = m >> 11;               // /2048
      const int s = m & (S_ - 1);
      const bf16_t v = (bf16_t)(acc[j][r] + bb);
      if (TRANS_OUT)
        Out[(((size_t)b * H_ + h) * DK_ + d) * S_ + s] = v;
      else
        Out[(((size_t)b * H_ + h) * S_ + s) * DK_ + d] = v;
    }
  }
}

// ---------------------------------------------------------------------------
// Kernel 2: flash attention, one wave per (b, h, 16-query tile)
// K/V/src_mask tiles staged Global->LDS by the TDM, double buffered.
// grid = B*H*(S/16) = 8192
// ---------------------------------------------------------------------------
__global__ __launch_bounds__(32) void flash_attn(
    const bf16_t* __restrict__ Q,    // [B,H,S,DK]
    const bf16_t* __restrict__ Kt,   // [B,H,S,DK]
    const bf16_t* __restrict__ Vt,   // [B,H,DK,S]  (transposed)
    const int*    __restrict__ pad,  // [B,S]
    const float*  __restrict__ srcm, // [S,S]
    bf16_t* __restrict__ O)          // [B,S,H*DK] bf16
{
  __shared__ __align__(128) bf16_t kbuf[2][32 * DK_];  // 32 keys x 64 d
  __shared__ __align__(128) bf16_t vbuf[2][DK_ * 32];  // 64 d x 32 keys
  __shared__ __align__(128) float  mbuf[2][16 * 32];   // 16 q x 32 keys src_mask
  __shared__ __align__(128) bf16_t pbuf[16 * 32];      // P tile bounce

  const int blk = blockIdx.x;
  const int qt  = blk & (S_ / 16 - 1);     // 128 q-tiles
  const int bh  = blk >> 7;
  const int h   = bh & (H_ - 1);
  const int b   = bh >> 4;

  const size_t bh_off = ((size_t)b * H_ + h) * S_ * DK_;
  const bf16_t* Kbase = Kt + bh_off;                 // [S, 64]
  const bf16_t* Vbase = Vt + bh_off;                 // [64, S]
  const bf16_t* Qb    = Q  + bh_off + (size_t)qt * 16 * DK_;
  const float*  Mbase = srcm + (size_t)(qt * 16) * S_;  // 16 mask rows

  // kick off TDM loads for the first 32-key step
  tdm_load_2d(lds_addr_of(&kbuf[0][0]), Kbase, 1, DK_, S_, DK_, 32, DK_);
  tdm_load_2d(lds_addr_of(&vbuf[0][0]), Vbase, 1, S_, DK_, 32, DK_, S_);
  tdm_load_2d(lds_addr_of(&mbuf[0][0]), Mbase, 2, S_, 16, 32, 16, S_);

  const v16bf a0 = load_a_bf16(Qb,      DK_);   // d 0..31
  const v16bf a1 = load_a_bf16(Qb + 32, DK_);   // d 32..63

  const int lane  = threadIdx.x & 31;
  const int nlo   = lane & 15;
  const int rbase = (lane >> 4) << 3;           // rows rbase..rbase+7

  float row_m[8], row_s[8];
#pragma unroll
  for (int r = 0; r < 8; ++r) { row_m[r] = -1e30f; row_s[r] = 0.0f; }
  v8f acc[4] = {};

  const float scale = 0.125f;                   // 1/sqrt(64)

  int cur = 0;
  for (int kt = 0; kt < S_; kt += 32) {
    // prefetch next step via TDM into the other buffers, then wait for current
    if (kt + 32 < S_) {
      tdm_load_2d(lds_addr_of(&kbuf[cur ^ 1][0]),
                  Kbase + (size_t)(kt + 32) * DK_, 1, DK_, S_, DK_, 32, DK_);
      tdm_load_2d(lds_addr_of(&vbuf[cur ^ 1][0]),
                  Vbase + (kt + 32), 1, S_, DK_, 32, DK_, S_);
      tdm_load_2d(lds_addr_of(&mbuf[cur ^ 1][0]),
                  Mbase + (kt + 32), 2, S_, 16, 32, 16, S_);
      __builtin_amdgcn_s_wait_tensorcnt(3);   // 3 newest still in flight
    } else {
      __builtin_amdgcn_s_wait_tensorcnt(0);
    }
    asm volatile("" ::: "memory");

    const bf16_t* kb_l = &kbuf[cur][0];
    const bf16_t* vb_l = &vbuf[cur][0];
    const float*  mb_l = &mbuf[cur][0];

    float sv[2][8];
#pragma unroll
    for (int sub = 0; sub < 2; ++sub) {
      const int kb = kt + sub * 16;
      const bf16_t* Kl = kb_l + sub * 16 * DK_;
      v16bf b0 = load_bT_bf16(Kl,      DK_);    // d 0..31
      v16bf b1 = load_bT_bf16(Kl + 32, DK_);    // d 32..63
      v8f s = {};
      s = wmma_bf16(a0, b0, s);
      s = wmma_bf16(a1, b1, s);
      const int kcol = kb + nlo;
      const float madd = (pad[b * S_ + kcol] == 0) ? -1.0e9f : 0.0f;
#pragma unroll
      for (int r = 0; r < 8; ++r) {
        sv[sub][r] = s[r] * scale + madd +
                     mb_l[(rbase + r) * 32 + sub * 16 + nlo];
      }
    }

    // online softmax: row max / sum across the 16 lanes holding each row
    float tmax[8];
#pragma unroll
    for (int r = 0; r < 8; ++r) tmax[r] = fmaxf(sv[0][r], sv[1][r]);
#pragma unroll
    for (int m = 1; m <= 8; m <<= 1)
#pragma unroll
      for (int r = 0; r < 8; ++r)
        tmax[r] = fmaxf(tmax[r], __shfl_xor(tmax[r], m, 32));

    float p0[8], p1[8], psum[8], esc[8];
#pragma unroll
    for (int r = 0; r < 8; ++r) {
      const float nm = fmaxf(row_m[r], tmax[r]);
      esc[r] = __expf(row_m[r] - nm);
      p0[r]  = __expf(sv[0][r] - nm);
      p1[r]  = __expf(sv[1][r] - nm);
      psum[r] = p0[r] + p1[r];
      row_m[r] = nm;
    }
#pragma unroll
    for (int m = 1; m <= 8; m <<= 1)
#pragma unroll
      for (int r = 0; r < 8; ++r)
        psum[r] += __shfl_xor(psum[r], m, 32);
#pragma unroll
    for (int r = 0; r < 8; ++r) row_s[r] = row_s[r] * esc[r] + psum[r];
#pragma unroll
    for (int j = 0; j < 4; ++j)
#pragma unroll
      for (int r = 0; r < 8; ++r) acc[j][r] *= esc[r];

    // P tile (C-layout) -> LDS
#pragma unroll
    for (int r = 0; r < 8; ++r) {
      pbuf[(rbase + r) * 32 + nlo]      = (bf16_t)p0[r];
      pbuf[(rbase + r) * 32 + 16 + nlo] = (bf16_t)p1[r];
    }
    __syncthreads();

    // P tile back as A-layout (16x32); V B-tiles are contiguous rows of vbuf
    const v16bf pa = load_a_bf16(pbuf, 32);
#pragma unroll
    for (int j = 0; j < 4; ++j) {
      v16bf vb = load_bT_bf16(vb_l + (size_t)j * 16 * 32, 32);  // d rows j*16..
      acc[j] = wmma_bf16(pa, vb, acc[j]);
    }
    __syncthreads();
    cur ^= 1;
  }

  // epilogue: normalize and store to [B,S,H*DK]
#pragma unroll
  for (int j = 0; j < 4; ++j) {
#pragma unroll
    for (int r = 0; r < 8; ++r) {
      const int qrow = qt * 16 + rbase + r;
      const int d    = j * 16 + nlo;
      const float o  = acc[j][r] / row_s[r];
      O[((size_t)b * S_ + qrow) * DM_ + h * DK_ + d] = (bf16_t)o;
    }
  }
}

// ---------------------------------------------------------------------------
// Kernel 3: out = A(bf16) @ Wo^T + bo, f32 output [M_, DM_]
// ---------------------------------------------------------------------------
__global__ __launch_bounds__(32) void proj_out_kernel(
    const bf16_t* __restrict__ A,    // [M_, DM_] bf16
    const float*  __restrict__ W,    // [DM_, DM_]
    const float*  __restrict__ bias, // [DM_]
    float* __restrict__ Out)         // [M_, DM_]
{
  const int tile = blockIdx.x;
  const int m0 = (tile >> 4) * 16;
  const int n0 = (tile & 15) * 64;

  v8f acc[4] = {};
  for (int k0 = 0; k0 < DM_; k0 += 32) {
    v16bf a = load_a_bf16(A + (size_t)m0 * DM_ + k0, DM_);
#pragma unroll
    for (int j = 0; j < 4; ++j) {
      v16bf bt = load_bT_f32(W + (size_t)(n0 + j * 16) * DM_ + k0, DM_);
      acc[j] = wmma_bf16(a, bt, acc[j]);
    }
  }

  const int lane  = threadIdx.x & 31;
  const int nlo   = lane & 15;
  const int rbase = (lane >> 4) << 3;
#pragma unroll
  for (int j = 0; j < 4; ++j) {
    const int n = n0 + j * 16 + nlo;
    const float bb = bias[n];
#pragma unroll
    for (int r = 0; r < 8; ++r) {
      Out[(size_t)(m0 + rbase + r) * DM_ + n] = acc[j][r] + bb;
    }
  }
}

// ---------------------------------------------------------------------------
extern "C" void kernel_launch(void* const* d_in, const int* in_sizes, int n_in,
                              void* d_out, int out_size, void* d_ws, size_t ws_size,
                              hipStream_t stream) {
  (void)in_sizes; (void)n_in; (void)out_size; (void)ws_size;

  const float* query  = (const float*)d_in[0];
  const float* key_in = (const float*)d_in[1];
  const float* value  = (const float*)d_in[2];
  const float* srcm   = (const float*)d_in[3];
  const int*   pad    = (const int*)d_in[4];
  const float* Wq = (const float*)d_in[5];
  const float* bq = (const float*)d_in[6];
  const float* Wk = (const float*)d_in[7];
  const float* bk = (const float*)d_in[8];
  const float* Wv = (const float*)d_in[9];
  const float* bv = (const float*)d_in[10];
  const float* Wo = (const float*)d_in[11];
  const float* bo = (const float*)d_in[12];
  float* out = (float*)d_out;

  const size_t elems = (size_t)B_ * H_ * S_ * DK_;   // 8M bf16 per buffer
  bf16_t* qb = (bf16_t*)d_ws;
  bf16_t* kb = qb + elems;
  bf16_t* vb = kb + elems;    // V stored as [B,H,DK,S]
  bf16_t* ab = vb + elems;

  const int gemm_grid = (M_ / 16) * (DM_ / 64);      // 8192
  const int attn_grid = B_ * H_ * (S_ / 16);         // 8192

  proj_split_heads<false><<<gemm_grid, 32, 0, stream>>>(query,  Wq, bq, qb);
  proj_split_heads<false><<<gemm_grid, 32, 0, stream>>>(key_in, Wk, bk, kb);
  proj_split_heads<true ><<<gemm_grid, 32, 0, stream>>>(value,  Wv, bv, vb);
  flash_attn<<<attn_grid, 32, 0, stream>>>(qb, kb, vb, pad, srcm, ab);
  proj_out_kernel<<<gemm_grid, 32, 0, stream>>>(ab, Wo, bo, out);
}
